// CoarseToFineCursorDecoder2d_71691594105198
// MI455X (gfx1250) — matrix operand order repaired and checked by reading, not compiled, not run on hardware
//
#include <hip/hip_runtime.h>
#include <math.h>

// ---------------------------------------------------------------------------
// CoarseToFineCursorDecoder2d for MI455X (gfx1250, wave32, WMMA)
//
// Roofline: ~21 GFLOP of GEMM vs ~537MB of output stores -> output-bound
// (~23us @ 23.3TB/s). Strategy: bf16 v_wmma_f32_16x16x32_bf16 for all six
// GEMMs (f32 accumulate), fused scatter+permute epilogue writing d_out once.
// ---------------------------------------------------------------------------

#define BATCH 1024
#define CDIM  768
#define NPOS  256
#define FDIM  512
#define KTOP  4
#define OUTSTRIDE 131073           // 1 + NPOS*FDIM
#define LOG_F 6.2383246250395077f  // log(512)

typedef __attribute__((ext_vector_type(8)))  unsigned short v8u;
typedef __attribute__((ext_vector_type(16))) unsigned short v16u;
typedef __attribute__((ext_vector_type(16))) __bf16        v16bf;
typedef __attribute__((ext_vector_type(8)))  float         v8f;
typedef unsigned short u16;

__device__ __forceinline__ u16 f2bf(float f) {
  unsigned int u = __float_as_uint(f);
  unsigned int r = u + 0x7FFFu + ((u >> 16) & 1u);   // round-to-nearest-even
  return (u16)(r >> 16);
}

__device__ __forceinline__ float gelu_f(float x) {
  // tanh approximation (jax.nn.gelu default)
  float x3 = x * x * x;
  return 0.5f * x * (1.0f + tanhf(0.7978845608028654f * (x + 0.044715f * x3)));
}

// --------------------------- weight convert ---------------------------------
// W (Kd x Nd) f32 row-major  ->  Wt (Nd x Kd) bf16 row-major (transposed so a
// WMMA B-column is contiguous; each lane reads it as two 16B loads).
__global__ void conv_wt_kernel(const float* __restrict__ W, u16* __restrict__ Wt,
                               int Kd, int Nd) {
  int i = blockIdx.x * blockDim.x + threadIdx.x;
  int total = Kd * Nd;
  if (i >= total) return;
  int n = i / Kd, k = i - n * Kd;
  Wt[(size_t)n * Kd + k] = f2bf(W[(size_t)k * Nd + n]);
}

// --------------------- LayerNorm(x) + no-op head ----------------------------
// One 256-thread block per batch row. Emits bf16 xn and out[b*OUTSTRIDE].
__global__ void ln_noop_kernel(const float* __restrict__ x,
                               const float* __restrict__ g,
                               const float* __restrict__ bvec,
                               const float* __restrict__ nW,
                               const float* __restrict__ nb,
                               u16* __restrict__ xn_bf,
                               float* __restrict__ out) {
  __shared__ float red[256];
  const int b = blockIdx.x, tid = threadIdx.x;
  const float* xr = x + (size_t)b * CDIM;

  float s = 0.f;
  for (int j = 0; j < 3; ++j) s += xr[tid + 256 * j];
  red[tid] = s; __syncthreads();
  for (int st = 128; st > 0; st >>= 1) { if (tid < st) red[tid] += red[tid + st]; __syncthreads(); }
  const float mean = red[0] * (1.0f / CDIM); __syncthreads();

  float vs = 0.f;
  for (int j = 0; j < 3; ++j) { float d = xr[tid + 256 * j] - mean; vs += d * d; }
  red[tid] = vs; __syncthreads();
  for (int st = 128; st > 0; st >>= 1) { if (tid < st) red[tid] += red[tid + st]; __syncthreads(); }
  const float rstd = rsqrtf(red[0] * (1.0f / CDIM) + 1e-5f); __syncthreads();

  float dot = 0.f;
  for (int j = 0; j < 3; ++j) {
    int c = tid + 256 * j;
    float xv = (xr[c] - mean) * rstd * g[c] + bvec[c];
    xn_bf[(size_t)b * CDIM + c] = f2bf(xv);
    dot += xv * nW[c];
  }
  red[tid] = dot; __syncthreads();
  for (int st = 128; st > 0; st >>= 1) { if (tid < st) red[tid] += red[tid + st]; __syncthreads(); }
  if (tid == 0) out[(size_t)b * OUTSTRIDE] = red[0] + nb[0];
}

// ------------------------------ WMMA GEMM -----------------------------------
// out(M x Nd) = act(A(M x Kd)bf16 @ W(Kd x Nd)bf16 + bias), W given transposed.
// One wave per 16x16 output tile, K-loop in steps of 32 via
// v_wmma_f32_16x16x32_bf16. ISA 16-bit A layout (16x32, wave32):
//   lane half selects K += 8; elems 0..7 -> K=kb+off..+7, 8..15 -> K=kb+16+off..
// B layout (32x16): lane%16 = column, same K mapping as A (half selects K += 8).
template <bool GELU, bool OUTBF>
__global__ void gemm_bf16_wmma(const u16* __restrict__ A,
                               const u16* __restrict__ Wt,
                               const float* __restrict__ bias,
                               void* __restrict__ outp,
                               int Kd, int Nd) {
  const int lane = threadIdx.x & 31;
  const int wave = threadIdx.x >> 5;
  const int Nt   = Nd >> 4;
  const int t    = blockIdx.x * 8 + wave;   // grid sized exactly: no guard, EXEC all-1s
  const int tM = t / Nt, tN = t % Nt;
  const int mrow = tM * 16 + (lane & 15);
  const int ncol = tN * 16 + (lane & 15);
  const int aoff = (lane >= 16) ? 8 : 0;
  const int boff = (lane >= 16) ? 16 : 0;
  const u16* __restrict__ Arow = A + (size_t)mrow * Kd;
  const u16* __restrict__ Wrow = Wt + (size_t)ncol * Kd;

  v8f acc = {};
  for (int kb = 0; kb < Kd; kb += 32) {
    v8u alo = *(const v8u*)(Arow + kb + aoff);
    v8u ahi = *(const v8u*)(Arow + kb + aoff + 16);
    v16u av = __builtin_shufflevector(alo, ahi, 0, 1, 2, 3, 4, 5, 6, 7,
                                      8, 9, 10, 11, 12, 13, 14, 15);
    v8u blo = *(const v8u*)(Wrow + kb + aoff);
    v8u bhi = *(const v8u*)(Wrow + kb + aoff + 16);
    v16u bv = __builtin_shufflevector(blo, bhi, 0, 1, 2, 3, 4, 5, 6, 7,
                                      8, 9, 10, 11, 12, 13, 14, 15);
    acc = __builtin_amdgcn_wmma_f32_16x16x32_bf16(
        false, __builtin_bit_cast(v16bf, av),
        false, __builtin_bit_cast(v16bf, bv),
        (short)0, acc, false, false);
  }

  const float bcol = bias[ncol];
  const int rbase = tM * 16 + ((lane >= 16) ? 8 : 0);
  for (int r = 0; r < 8; ++r) {
    float v = acc[r] + bcol;
    if (GELU) v = gelu_f(v);
    size_t o = (size_t)(rbase + r) * Nd + ncol;
    if (OUTBF) ((u16*)outp)[o] = f2bf(v);
    else       ((float*)outp)[o] = v;
  }
}

// ------------------------------- top-4 --------------------------------------
// One thread per batch row; 4-pass masked argmax (matches top_k tie order).
__global__ void top4_kernel(const float* __restrict__ coarse, int* __restrict__ idx) {
  int b = blockIdx.x * blockDim.x + threadIdx.x;
  if (b >= BATCH) return;
  const float* row = coarse + (size_t)b * NPOS;
  int sel0 = -1, sel1 = -1, sel2 = -1, sel3 = -1;
  for (int k = 0; k < KTOP; ++k) {
    int best = -1; float bv = -3.4e38f;
    for (int n = 0; n < NPOS; ++n) {
      if (n == sel0 || n == sel1 || n == sel2 || n == sel3) continue;
      float v = row[n];
      if (v > bv) { bv = v; best = n; }
    }
    if (k == 0) sel0 = best; else if (k == 1) sel1 = best;
    else if (k == 2) sel2 = best; else sel3 = best;
    idx[b * KTOP + k] = best;
  }
}

// -------------------- build fine_in = [xn ; LN(emb[i])] ---------------------
// One 256-thread block per (b,k) routed row.
__global__ void fine_in_kernel(const u16* __restrict__ xn_bf,
                               const float* __restrict__ emb,
                               const float* __restrict__ g_e,
                               const float* __restrict__ b_e,
                               const int* __restrict__ idx,
                               u16* __restrict__ fine_in) {
  __shared__ float red[256];
  const int row = blockIdx.x;              // 0..4095
  const int b = row >> 2, tid = threadIdx.x;
  const int i = idx[row];
  const float* er = emb + (size_t)i * CDIM;

  float s = 0.f;
  for (int j = 0; j < 3; ++j) s += er[tid + 256 * j];
  red[tid] = s; __syncthreads();
  for (int st = 128; st > 0; st >>= 1) { if (tid < st) red[tid] += red[tid + st]; __syncthreads(); }
  const float mean = red[0] * (1.0f / CDIM); __syncthreads();

  float vs = 0.f;
  for (int j = 0; j < 3; ++j) { float d = er[tid + 256 * j] - mean; vs += d * d; }
  red[tid] = vs; __syncthreads();
  for (int st = 128; st > 0; st >>= 1) { if (tid < st) red[tid] += red[tid + st]; __syncthreads(); }
  const float rstd = rsqrtf(red[0] * (1.0f / CDIM) + 1e-5f);

  u16* fr = fine_in + (size_t)row * (2 * CDIM);
  for (int j = 0; j < 3; ++j) {
    int c = tid + 256 * j;
    fr[c] = xn_bf[(size_t)b * CDIM + c];                       // broadcast xn
    float ev = (er[c] - mean) * rstd * g_e[c] + b_e[c];        // LN(emb[i])
    fr[CDIM + c] = f2bf(ev);
  }
}

// ------------------- upd = fine + logF - logsumexp(fine) --------------------
// One 256-thread block per routed row of 512 logits; rewrites in place.
__global__ void lse_upd_kernel(float* __restrict__ fine) {
  __shared__ float red[256];
  const int row = blockIdx.x, tid = threadIdx.x;
  float* fr = fine + (size_t)row * FDIM;
  float v0 = fr[tid], v1 = fr[tid + 256];

  red[tid] = fmaxf(v0, v1); __syncthreads();
  for (int st = 128; st > 0; st >>= 1) { if (tid < st) red[tid] = fmaxf(red[tid], red[tid + st]); __syncthreads(); }
  const float mx = red[0]; __syncthreads();

  red[tid] = __expf(v0 - mx) + __expf(v1 - mx); __syncthreads();
  for (int st = 128; st > 0; st >>= 1) { if (tid < st) red[tid] += red[tid + st]; __syncthreads(); }
  const float lse = mx + __logf(red[0]);

  const float shift = LOG_F - lse;
  fr[tid] = v0 + shift;
  fr[tid + 256] = v1 + shift;
}

// ---------------- fused scatter + permute + output write --------------------
// 1 block per (b,n). xfull permute (b,ch,cw,fh,fw,fp)->(b,ch,fh,cw,fw,fp):
// out off = 1 + ch*8192 + fh*512 + cw*32 + (fw*2+fp). Pure streaming stores.
__global__ void assemble_out_kernel(const float* __restrict__ coarse,
                                    const int* __restrict__ idx,
                                    const float* __restrict__ upd,
                                    float* __restrict__ out) {
  const int bn = blockIdx.x;
  const int b = bn >> 8;
  const int n = bn & 255;
  const float base = coarse[bn] - LOG_F;
  const int* ib = idx + b * KTOP;
  const float* u = nullptr;
  for (int j = 0; j < KTOP; ++j)
    if (ib[j] == n) u = upd + (size_t)(b * KTOP + j) * FDIM;
  const int ch = n >> 4, cw = n & 15;
  float* orow = out + (size_t)b * OUTSTRIDE + 1 + ch * 8192 + cw * 32;
  for (int f = threadIdx.x; f < FDIM; f += 256) {
    float v = base + (u ? u[f] : 0.0f);
    int fh = f >> 5, r = f & 31;           // f = fh*32 + fw*2 + fp
    orow[fh * 512 + r] = v;
  }
}

// ---------------------------------------------------------------------------
extern "C" void kernel_launch(void* const* d_in, const int* in_sizes, int n_in,
                              void* d_out, int out_size, void* d_ws, size_t ws_size,
                              hipStream_t stream) {
  const float* x    = (const float*)d_in[0];
  const float* g_in = (const float*)d_in[1];
  const float* b_in = (const float*)d_in[2];
  const float* cW1  = (const float*)d_in[3];  const float* cb1 = (const float*)d_in[4];
  const float* cW2  = (const float*)d_in[5];  const float* cb2 = (const float*)d_in[6];
  const float* cW3  = (const float*)d_in[7];  const float* cb3 = (const float*)d_in[8];
  const float* emb  = (const float*)d_in[9];
  const float* g_e  = (const float*)d_in[10]; const float* b_e = (const float*)d_in[11];
  const float* fW1  = (const float*)d_in[12]; const float* fb1 = (const float*)d_in[13];
  const float* fW2  = (const float*)d_in[14]; const float* fb2 = (const float*)d_in[15];
  const float* fW3  = (const float*)d_in[16]; const float* fb3 = (const float*)d_in[17];
  const float* nW   = (const float*)d_in[18]; const float* nb  = (const float*)d_in[19];
  float* out = (float*)d_out;

  // ---- workspace layout (256B aligned) ----
  char* base = (char*)d_ws;
  size_t off = 0;
  auto alloc = [&](size_t bytes) -> char* {
    char* p = base + off;
    off = (off + bytes + 255) & ~(size_t)255;
    return p;
  };
  u16*   xn_bf   = (u16*)  alloc((size_t)BATCH * CDIM * 2);
  u16*   cW1t    = (u16*)  alloc((size_t)CDIM * CDIM * 2);
  u16*   cW2t    = (u16*)  alloc((size_t)CDIM * CDIM * 2);
  u16*   cW3t    = (u16*)  alloc((size_t)CDIM * NPOS * 2);
  u16*   fW1t    = (u16*)  alloc((size_t)2 * CDIM * CDIM * 2);
  u16*   fW2t    = (u16*)  alloc((size_t)CDIM * CDIM * 2);
  u16*   fW3t    = (u16*)  alloc((size_t)CDIM * FDIM * 2);
  u16*   h1      = (u16*)  alloc((size_t)BATCH * CDIM * 2);
  u16*   h2      = (u16*)  alloc((size_t)BATCH * CDIM * 2);
  float* coarse  = (float*)alloc((size_t)BATCH * NPOS * 4);
  int*   idx     = (int*)  alloc((size_t)BATCH * KTOP * 4);
  u16*   fine_in = (u16*)  alloc((size_t)BATCH * KTOP * 2 * CDIM * 2);
  u16*   f1      = (u16*)  alloc((size_t)BATCH * KTOP * CDIM * 2);
  u16*   f2      = (u16*)  alloc((size_t)BATCH * KTOP * CDIM * 2);
  float* fine    = (float*)alloc((size_t)BATCH * KTOP * FDIM * 4);

  // ---- weight convert + transpose to bf16 ----
  auto conv = [&](const float* W, u16* Wt, int Kd, int Nd) {
    int total = Kd * Nd;
    conv_wt_kernel<<<(total + 255) / 256, 256, 0, stream>>>(W, Wt, Kd, Nd);
  };
  conv(cW1, cW1t, CDIM, CDIM);
  conv(cW2, cW2t, CDIM, CDIM);
  conv(cW3, cW3t, CDIM, NPOS);
  conv(fW1, fW1t, 2 * CDIM, CDIM);
  conv(fW2, fW2t, CDIM, CDIM);
  conv(fW3, fW3t, CDIM, FDIM);

  // ---- LN + no-op head ----
  ln_noop_kernel<<<BATCH, 256, 0, stream>>>(x, g_in, b_in, nW, nb, xn_bf, out);

  // ---- coarse MLP3 (WMMA): tiles = (M/16)*(Nd/16), 8 waves/block ----
  gemm_bf16_wmma<true,  true ><<<(BATCH/16)*(CDIM/16)/8, 256, 0, stream>>>(xn_bf, cW1t, cb1, h1,     CDIM, CDIM);
  gemm_bf16_wmma<true,  true ><<<(BATCH/16)*(CDIM/16)/8, 256, 0, stream>>>(h1,    cW2t, cb2, h2,     CDIM, CDIM);
  gemm_bf16_wmma<false, false><<<(BATCH/16)*(NPOS/16)/8, 256, 0, stream>>>(h2,    cW3t, cb3, coarse, CDIM, NPOS);

  // ---- routing ----
  top4_kernel<<<(BATCH + 255) / 256, 256, 0, stream>>>(coarse, idx);
  fine_in_kernel<<<BATCH * KTOP, 256, 0, stream>>>(xn_bf, emb, g_e, b_e, idx, fine_in);

  // ---- fine MLP3 (WMMA) ----
  const int MF = BATCH * KTOP;  // 4096 routed rows
  gemm_bf16_wmma<true,  true ><<<(MF/16)*(CDIM/16)/8, 256, 0, stream>>>(fine_in, fW1t, fb1, f1,   2 * CDIM, CDIM);
  gemm_bf16_wmma<true,  true ><<<(MF/16)*(CDIM/16)/8, 256, 0, stream>>>(f1,      fW2t, fb2, f2,   CDIM,     CDIM);
  gemm_bf16_wmma<false, false><<<(MF/16)*(FDIM/16)/8, 256, 0, stream>>>(f2,      fW3t, fb3, fine, CDIM,     FDIM);

  // ---- logsumexp update + fused scatter/permute output ----
  lse_upd_kernel<<<MF, 256, 0, stream>>>(fine);
  assemble_out_kernel<<<BATCH * NPOS, 256, 0, stream>>>(coarse, idx, fine, out);
}
